// PC_Model_40638980555099
// MI455X (gfx1250) — matrix operand order repaired and checked by reading, not compiled, not run on hardware
//
#include <hip/hip_runtime.h>

// ---------------------------------------------------------------------------
// H0 Vietoris-Rips Betti curves: [64,1024,16] f32 -> [64,1024] f32
//
//  Phase 0 (prep):      squared norms (f32) + bf16-converted points -> ws
//  Phase 1 (gram_dist): TDM-stage batch points into LDS (tensor_load_to_lds),
//                       D = sqrt(max(sq_i+sq_j-2*P.P^T, 1e-12)) via
//                       v_wmma_f32_16x16x32_bf16, diag=+inf, f32 -> ws
//  Phase 2 (prim):      register-resident Prim's MST (1023 serial steps,
//                       1 barrier/step), then Betti curve + normalization.
//
// Workspace layout (requires ws_size >= ~258.3 MB):
//   [0,           2MB)        : bf16 points, 64*1024*16 * 2B
//   [2MB,         2MB+256KB)  : squared norms, 64*1024 * 4B
//   [2MB+256KB,   +256MB)     : D matrix, 64*1024*1024 * 4B
// ---------------------------------------------------------------------------

typedef __attribute__((ext_vector_type(16))) __bf16        v16bf;
typedef __attribute__((ext_vector_type(8)))  float         v8f;
typedef __attribute__((ext_vector_type(8)))  unsigned int  v8u;
typedef __attribute__((ext_vector_type(4)))  unsigned int  v4u;
typedef __attribute__((ext_vector_type(4)))  unsigned int  u32x4;
typedef __attribute__((ext_vector_type(8)))  int           i32x8;
typedef __attribute__((ext_vector_type(4)))  int           i32x4;

#define NB 64
#define NS 1024
#define NC 16

__device__ __forceinline__ unsigned int pk_bf16(float lo, float hi) {
    // round-to-nearest-even f32 -> bf16, packed pair
    unsigned int a = __float_as_uint(lo);
    a = (a + 0x7FFFu + ((a >> 16) & 1u)) >> 16;
    unsigned int b = __float_as_uint(hi);
    b = (b + 0x7FFFu + ((b >> 16) & 1u)) >> 16;
    return a | (b << 16);
}

// ---------------- Phase 0: norms + bf16 conversion -------------------------
__global__ __launch_bounds__(256) void prep_kernel(
    const float* __restrict__ x, unsigned short* __restrict__ pbf,
    float* __restrict__ sq) {
    int gid = blockIdx.x * blockDim.x + threadIdx.x;  // point id in [0, NB*NS)
    const float4* xr = (const float4*)(x + (size_t)gid * NC);
    float4 a = xr[0], b = xr[1], c = xr[2], d = xr[3];
    float s = a.x * a.x + a.y * a.y + a.z * a.z + a.w * a.w +
              b.x * b.x + b.y * b.y + b.z * b.z + b.w * b.w +
              c.x * c.x + c.y * c.y + c.z * c.z + c.w * c.w +
              d.x * d.x + d.y * d.y + d.z * d.z + d.w * d.w;
    sq[gid] = s;
    uint4 p0, p1;
    p0.x = pk_bf16(a.x, a.y); p0.y = pk_bf16(a.z, a.w);
    p0.z = pk_bf16(b.x, b.y); p0.w = pk_bf16(b.z, b.w);
    p1.x = pk_bf16(c.x, c.y); p1.y = pk_bf16(c.z, c.w);
    p1.z = pk_bf16(d.x, d.y); p1.w = pk_bf16(d.z, d.w);
    uint4* dst = (uint4*)(pbf + (size_t)gid * NC);
    dst[0] = p0;
    dst[1] = p1;
}

// ---------------- Phase 1: TDM-staged WMMA Gram -> distance matrix ---------
// grid = (64 row-tiles, 64 batches), block = 256 (8 waves).
// Wave w computes col-tiles {w, w+8, ..., w+56} of a 16-row strip.
__global__ __launch_bounds__(256) void gram_dist_kernel(
    const unsigned short* __restrict__ pbf, const float* __restrict__ sq,
    float* __restrict__ Dm) {
    const int b   = blockIdx.y;
    const int rt  = blockIdx.x;           // 16-row tile index
    const int wid  = threadIdx.x >> 5;
    const int lane = threadIdx.x & 31;
    const int l16  = lane & 15;

    // Whole batch's bf16 point block (1024 x 16 bf16 = 32 KB) staged in LDS.
    __shared__ __align__(32) unsigned short s_pts[NS * NC];

    const unsigned short* pB  = pbf + (size_t)b * NS * NC;
    const float*          sqB = sq  + (size_t)b * NS;
    float*                DB  = Dm  + (size_t)b * NS * NS;

    // ---- Tensor Data Mover: global -> LDS, one DMA per workgroup ----------
    // D# describes a 1-D tensor of 8192 dwords (32 KB), tile == tensor.
    if (threadIdx.x == 0) {
        const unsigned long long ga = (unsigned long long)(uintptr_t)pB;
        // LDS aperture: flat addr[31:0] == LDS byte offset.
        const unsigned lds_off = (unsigned)(uintptr_t)(&s_pts[0]);
        u32x4 g0;
        g0[0] = 1u;                                   // count=1, user mode
        g0[1] = lds_off;                              // lds_addr
        g0[2] = (unsigned)(ga & 0xFFFFFFFFull);       // global_addr[31:0]
        g0[3] = (unsigned)((ga >> 32) & 0x1FFFFFFull) // global_addr[56:32]
                | (2u << 30);                         // type=2 ("image")
        i32x8 g1;
        g1[0] = (int)(2u << 16);           // wg_mask=0, data_size=2 (4B)
        g1[1] = (int)(0x2000u << 16);      // tensor_dim0[15:0]=8192
        g1[2] = (int)(1u << 16);           // tensor_dim0[31:16]=0, tensor_dim1=1
        g1[3] = (int)(0x2000u << 16);      // tile_dim0=8192
        g1[4] = 1;                         // tile_dim1=1, tile_dim2=0
        g1[5] = 8192;                      // tensor_dim0_stride
        g1[6] = 0;
        g1[7] = 0;
        i32x4 z4 = {0, 0, 0, 0};
#if defined(__clang_major__) && (__clang_major__ >= 23)
        i32x8 z8 = {0, 0, 0, 0, 0, 0, 0, 0};
        __builtin_amdgcn_tensor_load_to_lds(g0, g1, z4, z4, z8, 0);
#else
        __builtin_amdgcn_tensor_load_to_lds(g0, g1, z4, z4, 0);
#endif
        __builtin_amdgcn_s_wait_tensorcnt(0);
    }
    __syncthreads();

    // A fragment: 16x32 bf16, only K=0..15 populated.
    // lanes 0-15 hold K=0..15 of row M=l16; lanes 16-31 (K=16..31) zero.
    const int arow  = rt * 16 + l16;
    const int kbase = (lane < 16) ? 0 : 8;
    v8u a8   = {0u, 0u, 0u, 0u, 0u, 0u, 0u, 0u};
    if (lane < 16) a8 = *(const v8u*)(&s_pts[arow * NC]);
    v16bf Af = __builtin_bit_cast(v16bf, a8);

    // Row norms for the 8 accumulator VGPRs (C/D layout: M = r or r+8).
    const int rowoff = (lane < 16) ? 0 : 8;
    float sqr[8];
#pragma unroll
    for (int r = 0; r < 8; ++r) sqr[r] = sqB[rt * 16 + rowoff + r];

    for (int ct = wid; ct < NS / 16; ct += 8) {
        // B fragment: 32x16 bf16, K=0..15 live in lanes 0-15 (full column),
        // lanes 16-31 (K=16..31) zero.
        v8u braw = {0u, 0u, 0u, 0u, 0u, 0u, 0u, 0u};
        if (lane < 16)
            braw = *(const v8u*)(&s_pts[(ct * 16 + l16) * NC]);
        v16bf Bf = __builtin_bit_cast(v16bf, braw);

        v8f acc = {0.f, 0.f, 0.f, 0.f, 0.f, 0.f, 0.f, 0.f};
        acc = __builtin_amdgcn_wmma_f32_16x16x32_bf16(
            /*neg_a=*/false, Af, /*neg_b=*/false, Bf,
            /*c_mod=*/(short)0, acc, /*reuse_a=*/false, /*reuse_b=*/false);

        const int   col = ct * 16 + l16;
        const float sqc = sqB[col];
#pragma unroll
        for (int r = 0; r < 8; ++r) {
            const int row = rt * 16 + rowoff + r;
            float d2 = fmaxf(sqr[r] + sqc - 2.0f * acc[r], 1e-12f);
            float dd = __builtin_amdgcn_sqrtf(d2);  // bare v_sqrt_f32
            if (row == col) dd = __builtin_inff();
            DB[(size_t)row * NS + col] = dd;
        }
    }
}

// ---------------- Phase 2: Prim's MST + Betti curve ------------------------
// One block (256 threads, 8 waves) per batch. Thread t owns points 4t..4t+3,
// mind[] lives in registers; 1 barrier per MST iteration.
__global__ __launch_bounds__(256) void prim_betti_kernel(
    const float* __restrict__ Dm, float* __restrict__ out) {
    const int b    = blockIdx.x;
    const int t    = threadIdx.x;
    const int wid  = t >> 5;
    const int lane = t & 31;
    const float* DB = Dm + (size_t)b * NS * NS;

    __shared__ float s_deaths[NS];  // deaths[0..1022]
    __shared__ float s_pv[16];      // double-buffered per-wave argmin values
    __shared__ int   s_pi[16];
    __shared__ float s_red[16];

    const float INF = __builtin_inff();

    // mind0 = D[0, :]; vertex 0 starts in tree.
    float4 m4 = ((const float4*)DB)[t];
    float mind[4] = {m4.x, m4.y, m4.z, m4.w};
    int treemask = 0;
    if (t == 0) { treemask |= 1; mind[0] = INF; }  // D[0][0] is inf anyway

    for (int it = 0; it < NS - 1; ++it) {
        // local argmin over 4 owned points (first-index tie-break = argmin)
        float bv = INF;
        int   bi = 0x7fffffff;
#pragma unroll
        for (int i = 0; i < 4; ++i) {
            float v = mind[i];
            if (v < bv) { bv = v; bi = 4 * t + i; }
        }
        // wave32 shuffle reduction
#pragma unroll
        for (int off = 16; off >= 1; off >>= 1) {
            float ov = __shfl_xor(bv, off, 32);
            int   oi = __shfl_xor(bi, off, 32);
            if (ov < bv || (ov == bv && oi < bi)) { bv = ov; bi = oi; }
        }
        const int buf = (it & 1) * 8;  // double buffer -> single barrier/iter
        if (lane == 0) { s_pv[buf + wid] = bv; s_pi[buf + wid] = bi; }
        __syncthreads();
        float w = INF;
        int   j = 0x7fffffff;
#pragma unroll
        for (int q = 0; q < 8; ++q) {
            float v  = s_pv[buf + q];
            int   qi = s_pi[buf + q];
            if (v < w || (v == w && qi < j)) { w = v; j = qi; }
        }
        if (t == 0) s_deaths[it] = w;

        // mind = where(in_tree, inf, min(mind, D[j]))
        const float4 rowv = ((const float4*)(DB + (size_t)j * NS))[t];
        float rv[4] = {rowv.x, rowv.y, rowv.z, rowv.w};
#pragma unroll
        for (int i = 0; i < 4; ++i) {
            const int p = 4 * t + i;
            if (p == j) { treemask |= (1 << i); mind[i] = INF; }
            else if (!(treemask & (1 << i))) mind[i] = fminf(mind[i], rv[i]);
        }
    }
    __syncthreads();

    // max death
    float mx = 0.f;
    for (int i = t; i < NS - 1; i += 256) mx = fmaxf(mx, s_deaths[i]);
#pragma unroll
    for (int off = 16; off >= 1; off >>= 1)
        mx = fmaxf(mx, __shfl_xor(mx, off, 32));
    if (lane == 0) s_red[wid] = mx;
    __syncthreads();
    float maxd = 0.f;
#pragma unroll
    for (int q = 0; q < 8; ++q) maxd = fmaxf(maxd, s_red[q]);

    // Betti curve: thread t owns bins 4t..4t+3
    const float scale = maxd / (float)(NS - 1);
    float tk[4], bt[4];
#pragma unroll
    for (int i = 0; i < 4; ++i) {
        tk[i] = scale * (float)(4 * t + i);
        bt[i] = 0.f;
    }
    for (int d = 0; d < NS - 1; ++d) {
        const float dd = s_deaths[d];  // broadcast LDS read
#pragma unroll
        for (int i = 0; i < 4; ++i) bt[i] += (dd > tk[i]) ? 1.f : 0.f;
    }

    // min / max of betti over all 1024 bins
    float lmin = fminf(fminf(bt[0], bt[1]), fminf(bt[2], bt[3]));
    float lmax = fmaxf(fmaxf(bt[0], bt[1]), fmaxf(bt[2], bt[3]));
#pragma unroll
    for (int off = 16; off >= 1; off >>= 1) {
        lmin = fminf(lmin, __shfl_xor(lmin, off, 32));
        lmax = fmaxf(lmax, __shfl_xor(lmax, off, 32));
    }
    __syncthreads();
    if (lane == 0) { s_red[wid] = lmax; s_red[8 + wid] = lmin; }
    __syncthreads();
    float hi = 0.f, lo = INF;
#pragma unroll
    for (int q = 0; q < 8; ++q) {
        hi = fmaxf(hi, s_red[q]);
        lo = fminf(lo, s_red[8 + q]);
    }
    const float inv = 1.0f / (hi - lo + 1e-8f);
#pragma unroll
    for (int i = 0; i < 4; ++i)
        out[(size_t)b * NS + 4 * t + i] = (bt[i] - lo) * inv;
}

// ---------------------------------------------------------------------------
extern "C" void kernel_launch(void* const* d_in, const int* in_sizes, int n_in,
                              void* d_out, int out_size, void* d_ws,
                              size_t ws_size, hipStream_t stream) {
    (void)in_sizes; (void)n_in; (void)out_size; (void)ws_size;
    const float* x = (const float*)d_in[0];
    float* out = (float*)d_out;

    char* ws = (char*)d_ws;
    unsigned short* pbf = (unsigned short*)ws;                          // 2 MB
    float* sq = (float*)(ws + (size_t)2 * 1024 * 1024);                 // 256 KB
    float* Dm = (float*)(ws + (size_t)2 * 1024 * 1024 + 256 * 1024);    // 256 MB

    prep_kernel<<<(NB * NS) / 256, 256, 0, stream>>>(x, pbf, sq);
    gram_dist_kernel<<<dim3(NS / 16, NB), 256, 0, stream>>>(pbf, sq, Dm);
    prim_betti_kernel<<<NB, 256, 0, stream>>>(Dm, out);
}
